// LRU_90297392431689
// MI455X (gfx1250) — matrix operand order repaired and checked by reading, not compiled, not run on hardware
//
#include <hip/hip_runtime.h>
#include <hip/hip_bf16.h>

// ---------------------------------------------------------------------------
// LRU layer for MI455X (gfx1250, wave32, WMMA, async global->LDS staging).
//   GEMM1: Bu = u @ [gamma*B_re ; gamma*B_im]^T       (bf16 WMMA, f32 accum)
//   scan : blocked diagonal complex scan over T        (f32)
//   GEMM2: y  = [x_re x_im u] @ [2C_re ; -2C_im ; D]^T (bf16 WMMA, f32 accum)
// ---------------------------------------------------------------------------

#define B_SZ   16
#define T_SZ   4096
#define IN_SZ  512
#define N_SZ   512
#define OUT_SZ 512
#define M_SZ   (B_SZ * T_SZ)          // 65536 rows
#define NCHUNK 32
#define CHLEN  (T_SZ / NCHUNK)        // 128

#define LDS_PITCH 72                  // 64 + 8 bf16 pad: spreads LDS banks

typedef __attribute__((ext_vector_type(16))) __bf16 v16bf;
typedef __attribute__((ext_vector_type(8)))  float  v8f;

union FragU {
    v16bf  v;
    float4 q[2];
};

__device__ __forceinline__ unsigned short f2bf(float f) {
    unsigned u = __float_as_uint(f);
    u += 0x7fffu + ((u >> 16) & 1u);          // round-to-nearest-even
    return (unsigned short)(u >> 16);
}

// Raw LDS byte offset = low 32 bits of the flat shared address (ISA aperture rule).
__device__ __forceinline__ unsigned lds_off(const void* p) {
    return (unsigned)(size_t)p;
}

__device__ __forceinline__ void async_copy_b128(unsigned lds, const void* gaddr) {
    asm volatile("global_load_async_to_lds_b128 %0, %1, off"
                 :: "v"(lds), "v"(gaddr) : "memory");
}

__device__ __forceinline__ void wait_async0() {
    asm volatile("s_wait_asynccnt 0x0" ::: "memory");
}

// ---------------------------------------------------------------------------
// prep_consts: Lambda (re,im), Lambda^CHLEN, gamma.  One block of 512 threads.
// ---------------------------------------------------------------------------
__global__ __launch_bounds__(512)
void prep_consts(const float* __restrict__ nu_log,
                 const float* __restrict__ theta_log,
                 const float* __restrict__ gamma_log,
                 float* __restrict__ LamRe, float* __restrict__ LamIm,
                 float* __restrict__ PowRe, float* __restrict__ PowIm,
                 float* __restrict__ Gam) {
    int n = threadIdx.x;
    float mod = expf(-expf(nu_log[n]));
    float th  = expf(theta_log[n]);
    float lr  = mod * cosf(th);
    float li  = mod * sinf(th);
    LamRe[n] = lr;
    LamIm[n] = li;
    Gam[n]   = expf(gamma_log[n]);
    float pr = 1.f, pi = 0.f;
    for (int i = 0; i < CHLEN; ++i) {
        float nr = pr * lr - pi * li;
        float ni = pr * li + pi * lr;
        pr = nr; pi = ni;
    }
    PowRe[n] = pr;
    PowIm[n] = pi;
}

// ---------------------------------------------------------------------------
// prep_w1: W1[n][i] = bf16(gamma[n]*B_re[n][i]); W1[n+512][i] = ...B_im...
// ---------------------------------------------------------------------------
__global__ __launch_bounds__(256)
void prep_w1(const float* __restrict__ B_re, const float* __restrict__ B_im,
             const float* __restrict__ Gam, unsigned short* __restrict__ W1) {
    int idx = blockIdx.x * blockDim.x + threadIdx.x;   // over 512*512
    int n = idx >> 9, i = idx & 511;
    float g = Gam[n];
    W1[(size_t)n * IN_SZ + i]          = f2bf(g * B_re[idx]);
    W1[(size_t)(n + N_SZ) * IN_SZ + i] = f2bf(g * B_im[idx]);
}

// ---------------------------------------------------------------------------
// prep_w2: W2[o][0:512]=2*C_re, [512:1024]=-2*C_im, [1024:1536]=D
// ---------------------------------------------------------------------------
__global__ __launch_bounds__(256)
void prep_w2(const float* __restrict__ C_re, const float* __restrict__ C_im,
             const float* __restrict__ D, unsigned short* __restrict__ W2) {
    int idx = blockIdx.x * blockDim.x + threadIdx.x;   // over 512*512
    int o = idx >> 9, k = idx & 511;
    size_t row = (size_t)o * 1536;
    W2[row + k]        = f2bf( 2.0f * C_re[idx]);
    W2[row + 512 + k]  = f2bf(-2.0f * C_im[idx]);
    W2[row + 1024 + k] = f2bf(D[idx]);
}

// ---------------------------------------------------------------------------
// prep_u: Xcat[m][1024 + i] = bf16(u[m][i])
// ---------------------------------------------------------------------------
__global__ __launch_bounds__(256)
void prep_u(const float* __restrict__ u, unsigned short* __restrict__ Xcat) {
    size_t idx = (size_t)blockIdx.x * blockDim.x + threadIdx.x; // over M*512
    size_t m = idx >> 9, i = idx & 511;
    Xcat[m * 1536 + 1024 + i] = f2bf(u[idx]);
}

// ---------------------------------------------------------------------------
// WMMA GEMM with async global->LDS double-buffered staging.
//   C[M, :] = A[M, K](bf16, row stride lda) @ W[Ncols, K]^T
// Block tile 128x128, K staged in 64-wide double-buffered LDS tiles.
// 8 waves, each wave a 32x64 tile = 2x4 16x16x32 WMMAs per K sub-step.
// Tile order swizzled: 8 M-tiles sweep all N-tiles (A slab stays in L2).
// ---------------------------------------------------------------------------
__global__ __launch_bounds__(256)
void gemm_bf16(const unsigned short* __restrict__ A, int lda,
               const unsigned short* __restrict__ W,
               float* __restrict__ C, int ldc, int K, int ntiles) {
    __shared__ unsigned short As[2][128 * LDS_PITCH];
    __shared__ unsigned short Bs[2][128 * LDS_PITCH];

    // L2-friendly tile order: supergroups of 8 M-tiles sweep every N-tile.
    const int bid   = blockIdx.x;
    const int super = bid / (8 * ntiles);
    const int rem   = bid % (8 * ntiles);
    const int mBlk  = (super * 8 + (rem % 8)) * 128;
    const int nBlk  = (rem / 8) * 128;

    const int tid   = threadIdx.x;
    const int lane  = tid & 31;
    const int wave  = tid >> 5;
    const int half  = lane >> 4;            // 0/1
    const int lr    = lane & 15;
    const int waveM = wave >> 1;            // 0..3 -> 32-row quadrant
    const int waveN = wave & 1;             // 0..1 -> 64-col half

    v8f acc[2][4] = {};

    const int NK = K >> 6;                  // number of 64-wide K stages

    auto stage = [&](int buf, int k0) {
        // A tile: 128 rows x 64 bf16 = 1024 x 16B chunks; 4 per thread.
#pragma unroll
        for (int it = 0; it < 4; ++it) {
            int ch  = tid + it * 256;
            int row = ch >> 3, cc = ch & 7;
            const unsigned short* g = A + (size_t)(mBlk + row) * lda + k0 + cc * 8;
            async_copy_b128(lds_off(&As[buf][row * LDS_PITCH + cc * 8]), g);
        }
        // B tile: 128 weight rows x 64 bf16.
#pragma unroll
        for (int it = 0; it < 4; ++it) {
            int ch  = tid + it * 256;
            int row = ch >> 3, cc = ch & 7;
            const unsigned short* g = W + (size_t)(nBlk + row) * K + k0 + cc * 8;
            async_copy_b128(lds_off(&Bs[buf][row * LDS_PITCH + cc * 8]), g);
        }
    };

    stage(0, 0);
    for (int ks = 0; ks < NK; ++ks) {
        const int cur = ks & 1;
        wait_async0();           // my async writes to buffer `cur` landed
        __syncthreads();         // everyone's writes landed; prev reads done
        if (ks + 1 < NK) stage(cur ^ 1, (ks + 1) << 6);

#pragma unroll
        for (int sub = 0; sub < 2; ++sub) {
            const int ko = sub * 32;
            FragU a[2];
#pragma unroll
            for (int mf = 0; mf < 2; ++mf) {
                const unsigned short* p =
                    &As[cur][(waveM * 32 + mf * 16 + lr) * LDS_PITCH + ko + half * 8];
                a[mf].q[0] = *(const float4*)(p);        // K +0..7  (or +8..15)
                a[mf].q[1] = *(const float4*)(p + 16);   // K +16..23 (or +24..31)
            }
            FragU b[4];
#pragma unroll
            for (int nf = 0; nf < 4; ++nf) {
                const unsigned short* p =
                    &Bs[cur][(waveN * 64 + nf * 16 + lr) * LDS_PITCH + ko + half * 8];
                b[nf].q[0] = *(const float4*)(p);        // K +0..7  (or +8..15)
                b[nf].q[1] = *(const float4*)(p + 16);   // K +16..23 (or +24..31)
            }
#pragma unroll
            for (int mf = 0; mf < 2; ++mf)
#pragma unroll
                for (int nf = 0; nf < 4; ++nf)
                    acc[mf][nf] = __builtin_amdgcn_wmma_f32_16x16x32_bf16(
                        false, a[mf].v, false, b[nf].v,
                        (short)0, acc[mf][nf], false, false);
        }
    }

#pragma unroll
    for (int mf = 0; mf < 2; ++mf)
#pragma unroll
        for (int nf = 0; nf < 4; ++nf)
#pragma unroll
            for (int v = 0; v < 8; ++v) {
                int row = mBlk + waveM * 32 + mf * 16 + v + half * 8;
                int col = nBlk + waveN * 64 + nf * 16 + lr;
                C[(size_t)row * ldc + col] = acc[mf][nf][v];
            }
}

// ---------------------------------------------------------------------------
// scan pass 1: per-(b, chunk, n) local scan with zero init -> final local state
// ---------------------------------------------------------------------------
__global__ __launch_bounds__(512)
void scan_local(const float* __restrict__ Bu,
                const float* __restrict__ LamRe, const float* __restrict__ LamIm,
                float* __restrict__ locRe, float* __restrict__ locIm) {
    int n = threadIdx.x;
    int bc = blockIdx.x;            // b * NCHUNK + c
    int b = bc / NCHUNK, c = bc % NCHUNK;
    float lr = LamRe[n], li = LamIm[n];
    float xr = 0.f, xi = 0.f;
    const float* p = Bu + ((size_t)(b * T_SZ + c * CHLEN)) * 1024 + n;
    for (int t = 0; t < CHLEN; ++t) {
        float br = p[0], bi = p[512];
        float nr = lr * xr - li * xi + br;
        float ni = lr * xi + li * xr + bi;
        xr = nr; xi = ni;
        p += 1024;
    }
    locRe[(size_t)bc * N_SZ + n] = xr;
    locIm[(size_t)bc * N_SZ + n] = xi;
}

// ---------------------------------------------------------------------------
// scan pass 2: carry sweep across chunks (stores state BEFORE each chunk)
// ---------------------------------------------------------------------------
__global__ __launch_bounds__(512)
void scan_carry(const float* __restrict__ locRe, const float* __restrict__ locIm,
                const float* __restrict__ PowRe, const float* __restrict__ PowIm,
                float* __restrict__ carRe, float* __restrict__ carIm) {
    int n = threadIdx.x;
    int b = blockIdx.x;
    float pr = PowRe[n], pi = PowIm[n];
    float cr = 0.f, ci = 0.f;
    for (int c = 0; c < NCHUNK; ++c) {
        size_t idx = ((size_t)b * NCHUNK + c) * N_SZ + n;
        carRe[idx] = cr;
        carIm[idx] = ci;
        float sr = locRe[idx], si = locIm[idx];
        float nr = pr * cr - pi * ci + sr;
        float ni = pr * ci + pi * cr + si;
        cr = nr; ci = ni;
    }
}

// ---------------------------------------------------------------------------
// scan pass 3: full scan with carry-in; emit x as bf16 into Xcat[:,0:1024]
// ---------------------------------------------------------------------------
__global__ __launch_bounds__(512)
void scan_final(const float* __restrict__ Bu,
                const float* __restrict__ LamRe, const float* __restrict__ LamIm,
                const float* __restrict__ carRe, const float* __restrict__ carIm,
                unsigned short* __restrict__ Xcat) {
    int n = threadIdx.x;
    int bc = blockIdx.x;
    int b = bc / NCHUNK, c = bc % NCHUNK;
    float lr = LamRe[n], li = LamIm[n];
    size_t cidx = (size_t)bc * N_SZ + n;
    float xr = carRe[cidx], xi = carIm[cidx];
    size_t row0 = (size_t)(b * T_SZ + c * CHLEN);
    const float* p = Bu + row0 * 1024 + n;
    unsigned short* q = Xcat + row0 * 1536 + n;
    for (int t = 0; t < CHLEN; ++t) {
        float br = p[0], bi = p[512];
        float nr = lr * xr - li * xi + br;
        float ni = lr * xi + li * xr + bi;
        xr = nr; xi = ni;
        q[0]   = f2bf(xr);
        q[512] = f2bf(xi);
        p += 1024;
        q += 1536;
    }
}

// ---------------------------------------------------------------------------
extern "C" void kernel_launch(void* const* d_in, const int* in_sizes, int n_in,
                              void* d_out, int out_size, void* d_ws, size_t ws_size,
                              hipStream_t stream) {
    const float* u         = (const float*)d_in[0];
    const float* nu_log    = (const float*)d_in[1];
    const float* theta_log = (const float*)d_in[2];
    const float* gamma_log = (const float*)d_in[3];
    const float* B_re      = (const float*)d_in[4];
    const float* B_im      = (const float*)d_in[5];
    const float* C_re      = (const float*)d_in[6];
    const float* C_im      = (const float*)d_in[7];
    const float* D         = (const float*)d_in[8];
    float* y               = (float*)d_out;

    char* ws = (char*)d_ws;
    size_t off = 0;
    auto carve = [&](size_t bytes) -> void* {
        void* p = ws + off;
        off += (bytes + 255) & ~(size_t)255;
        return p;
    };
    float* LamRe = (float*)carve(N_SZ * 4);
    float* LamIm = (float*)carve(N_SZ * 4);
    float* PowRe = (float*)carve(N_SZ * 4);
    float* PowIm = (float*)carve(N_SZ * 4);
    float* Gam   = (float*)carve(N_SZ * 4);
    unsigned short* W1   = (unsigned short*)carve((size_t)1024 * IN_SZ * 2);
    unsigned short* W2   = (unsigned short*)carve((size_t)OUT_SZ * 1536 * 2);
    float* locRe = (float*)carve((size_t)B_SZ * NCHUNK * N_SZ * 4);
    float* locIm = (float*)carve((size_t)B_SZ * NCHUNK * N_SZ * 4);
    float* carRe = (float*)carve((size_t)B_SZ * NCHUNK * N_SZ * 4);
    float* carIm = (float*)carve((size_t)B_SZ * NCHUNK * N_SZ * 4);
    unsigned short* Xcat = (unsigned short*)carve((size_t)M_SZ * 1536 * 2); // 192 MB
    float* Bu            = (float*)carve((size_t)M_SZ * 1024 * 4);          // 256 MB

    // --- prep ---
    prep_consts<<<1, 512, 0, stream>>>(nu_log, theta_log, gamma_log,
                                       LamRe, LamIm, PowRe, PowIm, Gam);
    prep_w1<<<(512 * 512) / 256, 256, 0, stream>>>(B_re, B_im, Gam, W1);
    prep_w2<<<(512 * 512) / 256, 256, 0, stream>>>(C_re, C_im, D, W2);
    prep_u<<<((size_t)M_SZ * 512) / 256, 256, 0, stream>>>(u, Xcat);

    // --- GEMM1: Bu[M,1024] = u_bf16 @ W1^T  (M/128=512 tiles, N/128=8 tiles) ---
    gemm_bf16<<<(M_SZ / 128) * 8, 256, 0, stream>>>(
        Xcat + 1024, 1536, W1, Bu, 1024, IN_SZ, 8);

    // --- blocked complex scan ---
    scan_local<<<B_SZ * NCHUNK, 512, 0, stream>>>(Bu, LamRe, LamIm, locRe, locIm);
    scan_carry<<<B_SZ, 512, 0, stream>>>(locRe, locIm, PowRe, PowIm, carRe, carIm);
    scan_final<<<B_SZ * NCHUNK, 512, 0, stream>>>(Bu, LamRe, LamIm, carRe, carIm, Xcat);

    // --- GEMM2: y[M,512] = Xcat @ W2^T  (512 M-tiles, 4 N-tiles) ---
    gemm_bf16<<<(M_SZ / 128) * 4, 256, 0, stream>>>(
        Xcat, 1536, W2, y, OUT_SZ, 1536, 4);
}
